// ReparamShiftDepthwiseBlock_72636486910482
// MI455X (gfx1250) — matrix operand order repaired and checked by reading, not compiled, not run on hardware
//
#include <hip/hip_runtime.h>

// ---- problem constants (from reference) ----
#define NB    64
#define CIN_  96
#define COUT_ 96
#define MID_  192
#define HDIM  56
#define HWSZ  3136   // 56*56
#define EPSB  1e-5f

typedef __attribute__((ext_vector_type(16))) __bf16 v16bf;
typedef __attribute__((ext_vector_type(8)))  float  v8f;
typedef __attribute__((ext_vector_type(4)))  float  v4f;
typedef __attribute__((ext_vector_type(4)))  unsigned int u32x4;

union FragU {
  v16bf v;
  u32x4 q[2];
  unsigned short u[16];
};

__device__ __forceinline__ unsigned short f32_to_bf16(float f) {
  union { float f; unsigned u; } x; x.f = f;
  unsigned r = x.u + 0x7fffu + ((x.u >> 16) & 1u);   // round-to-nearest-even
  return (unsigned short)(r >> 16);
}
__device__ __forceinline__ float bf16_to_f32(unsigned short h) {
  union { float f; unsigned u; } x; x.u = ((unsigned)h) << 16; return x.f;
}
__device__ __forceinline__ float silu_f(float r) {
  return r * (1.f / (1.f + __expf(-r)));
}

// A-fragment: 16x32 bf16 tile, row-major in LDS (row stride ld elements, row
// base 16B-aligned). This code's k order per lane (contiguous, as packed B) = two 8-element runs:
// lanes 0-15: K 0..15 of row (lane&15); lanes 16-31: K 16..31.
__device__ __forceinline__ v16bf frag_a(const unsigned short* p, int ld, int lane) {
  const unsigned short* row = p + (lane & 15) * ld + ((lane >> 4) & 1) * 16;
  FragU f;
  f.q[0] = *(const u32x4*)(row);        // K 0..7  (+16 upper half)
  f.q[1] = *(const u32x4*)(row + 8);    // K 8..15 (+16 upper half)
  return f.v;
}
// B-fragment from fragment-packed LDS: tile base p, lane's 16 bf16 contiguous.
__device__ __forceinline__ v16bf frag_b_packed(const unsigned short* p, int lane) {
  const unsigned short* src = p + lane * 16;
  FragU f;
  f.q[0] = *(const u32x4*)(src);
  f.q[1] = *(const u32x4*)(src + 8);
  return f.v;
}

// ---------------- prep: fold BN, pack weights into fragment order ----------
// Packed B layout: idx = ((ks*NSUB + ns)*32 + lanepos)*16 + (kk&15),
// lanepos = (n&15) + 16*(kk>>4), ks = k/32, kk = k%32, ns = n/16.
__global__ void k_prep(const float* __restrict__ expand_w, const float* __restrict__ expand_bn,
                       const float* __restrict__ dw_bn, const float* __restrict__ id_w,
                       const float* __restrict__ id_bn, const float* __restrict__ merge_bn,
                       const float* __restrict__ proj_w, const float* __restrict__ proj_bn,
                       unsigned short* __restrict__ wb1p, unsigned short* __restrict__ wb2p,
                       float* __restrict__ exp_s, float* __restrict__ exp_sh,
                       float* __restrict__ Ac, float* __restrict__ Bc, float* __restrict__ Cc,
                       float* __restrict__ proj_s, float* __restrict__ proj_sh) {
  int tid = threadIdx.x;
  for (int i = tid; i < MID_; i += 256) {
    float s = expand_bn[i] * rsqrtf(expand_bn[3 * MID_ + i] + EPSB);
    exp_s[i] = s;
    exp_sh[i] = expand_bn[MID_ + i] - expand_bn[2 * MID_ + i] * s;

    float sdw  = dw_bn[i] * rsqrtf(dw_bn[3 * MID_ + i] + EPSB);
    float shdw = dw_bn[MID_ + i] - dw_bn[2 * MID_ + i] * sdw;
    float sid  = id_bn[i] * rsqrtf(id_bn[3 * MID_ + i] + EPSB);
    float shid = id_bn[MID_ + i] - id_bn[2 * MID_ + i] * sid;
    float sm   = merge_bn[i] * rsqrtf(merge_bn[3 * MID_ + i] + EPSB);
    float shm  = merge_bn[MID_ + i] - merge_bn[2 * MID_ + i] * sm;
    Ac[i] = sdw * sm;                    // scales raw depthwise conv
    Bc[i] = id_w[i] * sid * sm;          // scales center value (identity branch)
    Cc[i] = (shdw + shid) * sm + shm;    // combined shift
  }
  for (int i = tid; i < COUT_; i += 256) {
    float s = proj_bn[i] * rsqrtf(proj_bn[3 * COUT_ + i] + EPSB);
    proj_s[i] = s;
    proj_sh[i] = proj_bn[COUT_ + i] - proj_bn[2 * COUT_ + i] * s;
  }
  for (int i = tid; i < CIN_ * MID_; i += 256) {       // expand weights, N=192 (12 nsub)
    int k = i / MID_, n = i % MID_;
    int ks = k >> 5, kk = k & 31, ns = n >> 4;
    int lanepos = (n & 15) + ((kk >> 4) << 4);
    wb1p[((ks * 12 + ns) * 32 + lanepos) * 16 + (kk & 15)] =
        f32_to_bf16(expand_w[n * CIN_ + k]);
  }
  for (int i = tid; i < MID_ * COUT_; i += 256) {      // proj weights, N=96 (6 nsub)
    int k = i / COUT_, n = i % COUT_;
    int ks = k >> 5, kk = k & 31, ns = n >> 4;
    int lanepos = (n & 15) + ((kk >> 4) << 4);
    wb2p[((ks * 6 + ns) * 32 + lanepos) * 16 + (kk & 15)] =
        f32_to_bf16(proj_w[n * MID_ + k]);
  }
}

// ---------------- GEMM1: 1x1 expand + BN + SiLU -> t (bf16) -----------------
// Per block: 64 spatial rows (one image), all 192 out channels, K=96.
__global__ __launch_bounds__(256) void k_expand(const float* __restrict__ x,
                                                const unsigned short* __restrict__ wb1p,
                                                const float* __restrict__ exp_s,
                                                const float* __restrict__ exp_sh,
                                                unsigned short* __restrict__ t_bf) {
  __shared__ unsigned short smem[25088];        // 50.2 KB
  unsigned short* As = smem;                    // [64][104]  (row = 208B, 16B-aligned)
  unsigned short* Bs = smem + 64 * 104;         // fragment-packed, 18432
  int tid = threadIdx.x;
  int tile = blockIdx.x;
  int b = tile / 49;
  int hw0 = (tile % 49) * 64;

  // A tile: pairs along K -> b32 LDS stores; coalesced f32 global loads.
  for (int i = tid; i < 64 * 48; i += 256) {
    int m = i & 63, k2 = i >> 6;                // k = 2*k2
    size_t base = ((size_t)b * CIN_ + 2 * k2) * HWSZ + hw0 + m;
    unsigned lo = f32_to_bf16(x[base]);
    unsigned hi = f32_to_bf16(x[base + HWSZ]);
    *(unsigned*)&As[m * 104 + 2 * k2] = lo | (hi << 16);
  }
  // B tile: straight b128 copy of packed weights.
  for (int i = tid; i < 18432 / 8; i += 256)
    ((u32x4*)Bs)[i] = ((const u32x4*)wb1p)[i];
  __syncthreads();

  int lane = tid & 31, wave = tid >> 5;
  int ms = wave >> 1, nh = wave & 1;            // 4 m-subtiles x (2 halves of 12 n-subtiles)
  v8f acc[6];
  for (int j = 0; j < 6; ++j)
    for (int e = 0; e < 8; ++e) acc[j][e] = 0.f;

#pragma unroll
  for (int ks = 0; ks < 3; ++ks) {
    v16bf a = frag_a(As + ms * 16 * 104 + ks * 32, 104, lane);
#pragma unroll
    for (int j = 0; j < 6; ++j) {
      int ns = nh * 6 + j;
      v16bf bb = frag_b_packed(Bs + ((ks * 12 + ns) << 9), lane);
      acc[j] = __builtin_amdgcn_wmma_f32_16x16x32_bf16(false, a, false, bb,
                                                       (short)0, acc[j], false, false);
    }
  }
  __syncthreads();

  unsigned short* Os = smem;                    // [192][72] staging (144B rows)
#pragma unroll
  for (int j = 0; j < 6; ++j) {
    int ns = nh * 6 + j;
    int n = ns * 16 + (lane & 15);
    float sc = exp_s[n], sh = exp_sh[n];
    int mbase = ms * 16 + 8 * (lane >> 4);
    FragU st;
#pragma unroll
    for (int v = 0; v < 8; ++v)
      st.u[v] = f32_to_bf16(silu_f(acc[j][v] * sc + sh));
    *(u32x4*)&Os[n * 72 + mbase] = st.q[0];     // one b128 store
  }
  __syncthreads();
  for (int i = tid; i < MID_ * 8; i += 256) {   // b128 global stores
    int n = i >> 3, m8 = (i & 7) * 8;
    u32x4 val = *(const u32x4*)&Os[n * 72 + m8];
    *(u32x4*)(t_bf + ((size_t)b * MID_ + n) * HWSZ + hw0 + m8) = val;
  }
}

// ------- depthwise 3x3 + identity + merged BN + SiLU + SE pool sums ---------
// shift+shuffle folded into addressing: shuffled[c] = t[(c%4)*48 + c/4] shifted
// by direction d = c%4 (0:up, 1:left, 2:down, 3:right).
__global__ __launch_bounds__(256) void k_dw(const unsigned short* __restrict__ t_bf,
                                            const float* __restrict__ dw_w,
                                            const float* __restrict__ Ac,
                                            const float* __restrict__ Bc,
                                            const float* __restrict__ Cc,
                                            unsigned short* __restrict__ u_bf,
                                            float* __restrict__ sesum) {
  __shared__ float plane[HWSZ];
  __shared__ float red[256];
  int tid = threadIdx.x;
  int bc = blockIdx.x;
  int b = bc / MID_, c = bc % MID_;
  int s = (c & 3) * 48 + (c >> 2);
  int d = c & 3;
  int dy = (d == 0) ? -1 : ((d == 2) ? 1 : 0);
  int dx = (d == 1) ? -1 : ((d == 3) ? 1 : 0);

  const unsigned short* src = t_bf + ((size_t)b * MID_ + s) * HWSZ;
  for (int i = tid; i < HWSZ / 8; i += 256) {   // b128 loads (8 bf16)
    FragU v;
    v.q[0] = ((const u32x4*)src)[i];
#pragma unroll
    for (int j = 0; j < 8; ++j) plane[i * 8 + j] = bf16_to_f32(v.u[j]);
  }
  float w[9];
#pragma unroll
  for (int i = 0; i < 9; ++i) w[i] = dw_w[c * 9 + i];
  float ac = Ac[c], bcc = Bc[c], cc = Cc[c];
  __syncthreads();

  float sum = 0.f;
  unsigned short* dst = u_bf + ((size_t)b * MID_ + c) * HWSZ;
  for (int p0 = tid * 2; p0 < HWSZ; p0 += 512) {  // 2 pixels/iter -> b32 stores
    unsigned short r01[2];
#pragma unroll
    for (int q = 0; q < 2; ++q) {
      int p = p0 + q;
      int h = p / HDIM, wq = p - h * HDIM;
      float dwv = 0.f, center = 0.f;
#pragma unroll
      for (int i = 0; i < 3; ++i) {
#pragma unroll
        for (int j = 0; j < 3; ++j) {
          int y = h + i - 1, xw = wq + j - 1;   // conv tap (conv zero-pad check)
          int ys = y + dy, xs = xw + dx;        // shifted source (shift zero check)
          float v = 0.f;
          if ((unsigned)y < HDIM && (unsigned)xw < HDIM &&
              (unsigned)ys < HDIM && (unsigned)xs < HDIM)
            v = plane[ys * HDIM + xs];
          dwv += w[i * 3 + j] * v;
          if (i == 1 && j == 1) center = v;
        }
      }
      float r = silu_f(dwv * ac + center * bcc + cc);
      sum += r;
      r01[q] = f32_to_bf16(r);
    }
    *(unsigned*)&dst[p0] = (unsigned)r01[0] | ((unsigned)r01[1] << 16);
  }
  red[tid] = sum;
  __syncthreads();
  for (int st = 128; st > 0; st >>= 1) {
    if (tid < st) red[tid] += red[tid + st];
    __syncthreads();
  }
  if (tid == 0) sesum[bc] = red[0];
}

// ---------------- grouped SE MLP -> sigmoid gate ----------------------------
__global__ void k_se(const float* __restrict__ sesum, const float* __restrict__ rw,
                     const float* __restrict__ rb, const float* __restrict__ ew,
                     const float* __restrict__ eb, float* __restrict__ gate) {
  __shared__ float mean[MID_];
  int b = blockIdx.x, c = threadIdx.x;          // 192 threads
  mean[c] = sesum[b * MID_ + c] * (1.f / (float)HWSZ);
  __syncthreads();
  int g = c >> 4;                               // 12 groups of 16
  float acc = eb[c];
#pragma unroll
  for (int o = 0; o < 2; ++o) {
    int r = g * 2 + o;
    float rv = rb[r];
#pragma unroll
    for (int i = 0; i < 16; ++i) rv += mean[g * 16 + i] * rw[r * 16 + i];
    rv = fmaxf(rv, 0.f);
    acc += rv * ew[c * 2 + o];
  }
  gate[b * MID_ + c] = 1.f / (1.f + __expf(-acc));
}

// -------- GEMM2: (u * gate) x proj_w + BN + residual + SiLU -> out ----------
__global__ __launch_bounds__(256) void k_project(const unsigned short* __restrict__ u_bf,
                                                 const float* __restrict__ gate,
                                                 const unsigned short* __restrict__ wb2p,
                                                 const float* __restrict__ proj_s,
                                                 const float* __restrict__ proj_sh,
                                                 const float* __restrict__ x,
                                                 float* __restrict__ out) {
  __shared__ unsigned short smem[31232];        // 62.5 KB
  unsigned short* As = smem;                    // [64][200] (row = 400B, 16B-aligned)
  unsigned short* Bs = smem + 64 * 200;         // fragment-packed, 18432
  int tid = threadIdx.x;
  int tile = blockIdx.x;
  int b = tile / 49;
  int hw0 = (tile % 49) * 64;

  for (int i = tid; i < 64 * 96; i += 256) {    // pairs along K -> b32 LDS stores
    int m = i & 63, k2 = i >> 6;
    int k = 2 * k2;
    size_t base = ((size_t)b * MID_ + k) * HWSZ + hw0 + m;
    unsigned lo = f32_to_bf16(bf16_to_f32(u_bf[base]) * gate[b * MID_ + k]);
    unsigned hi = f32_to_bf16(bf16_to_f32(u_bf[base + HWSZ]) * gate[b * MID_ + k + 1]);
    *(unsigned*)&As[m * 200 + k] = lo | (hi << 16);
  }
  for (int i = tid; i < 18432 / 8; i += 256)
    ((u32x4*)Bs)[i] = ((const u32x4*)wb2p)[i];
  __syncthreads();

  int lane = tid & 31, wave = tid >> 5;
  int ms = wave >> 1, nh = wave & 1;            // 4 m-subtiles x (2 halves of 6 n-subtiles)
  v8f acc[3];
  for (int j = 0; j < 3; ++j)
    for (int e = 0; e < 8; ++e) acc[j][e] = 0.f;

#pragma unroll
  for (int ks = 0; ks < 6; ++ks) {
    v16bf a = frag_a(As + ms * 16 * 200 + ks * 32, 200, lane);
#pragma unroll
    for (int j = 0; j < 3; ++j) {
      int ns = nh * 3 + j;
      v16bf bb = frag_b_packed(Bs + ((ks * 6 + ns) << 9), lane);
      acc[j] = __builtin_amdgcn_wmma_f32_16x16x32_bf16(false, a, false, bb,
                                                       (short)0, acc[j], false, false);
    }
  }
  __syncthreads();

  float* Os = (float*)smem;                     // [96][68] staging (272B rows)
#pragma unroll
  for (int j = 0; j < 3; ++j) {
    int ns = nh * 3 + j;
    int n = ns * 16 + (lane & 15);
    float sc = proj_s[n], sh = proj_sh[n];
    int mbase = ms * 16 + 8 * (lane >> 4);
    v4f t0, t1;
#pragma unroll
    for (int v = 0; v < 4; ++v) {
      t0[v] = acc[j][v] * sc + sh;
      t1[v] = acc[j][4 + v] * sc + sh;
    }
    *(v4f*)&Os[n * 68 + mbase] = t0;
    *(v4f*)&Os[n * 68 + mbase + 4] = t1;
  }
  __syncthreads();
  for (int i = tid; i < COUT_ * 16; i += 256) { // vectorized residual+silu+store
    int n = i >> 4, m4 = (i & 15) * 4;
    size_t gi = ((size_t)b * COUT_ + n) * HWSZ + hw0 + m4;
    v4f o = *(const v4f*)&Os[n * 68 + m4];
    v4f xr = *(const v4f*)&x[gi];
    v4f r;
#pragma unroll
    for (int e = 0; e < 4; ++e) r[e] = silu_f(o[e] + xr[e]);
    *(v4f*)&out[gi] = r;
  }
}

extern "C" void kernel_launch(void* const* d_in, const int* in_sizes, int n_in,
                              void* d_out, int out_size, void* d_ws, size_t ws_size,
                              hipStream_t stream) {
  (void)in_sizes; (void)n_in; (void)out_size; (void)ws_size;
  const float* x        = (const float*)d_in[0];
  const float* expand_w = (const float*)d_in[1];
  const float* expand_bn= (const float*)d_in[2];
  const float* dw_w     = (const float*)d_in[3];
  const float* dw_bn    = (const float*)d_in[4];
  const float* id_w     = (const float*)d_in[5];
  const float* id_bn    = (const float*)d_in[6];
  const float* merge_bn = (const float*)d_in[7];
  const float* se_red_w = (const float*)d_in[8];
  const float* se_red_b = (const float*)d_in[9];
  const float* se_exp_w = (const float*)d_in[10];
  const float* se_exp_b = (const float*)d_in[11];
  const float* proj_w   = (const float*)d_in[12];
  const float* proj_bn  = (const float*)d_in[13];
  float* out = (float*)d_out;

  char* ws = (char*)d_ws;
  size_t off = 0;
  auto take = [&](size_t bytes) -> char* {
    char* p = ws + off;
    off = (off + bytes + 255) & ~(size_t)255;
    return p;
  };
  unsigned short* t_bf  = (unsigned short*)take((size_t)NB * MID_ * HWSZ * 2);  // 77 MB
  unsigned short* u_bf  = (unsigned short*)take((size_t)NB * MID_ * HWSZ * 2);  // 77 MB
  unsigned short* wb1p  = (unsigned short*)take((size_t)CIN_ * MID_ * 2);
  unsigned short* wb2p  = (unsigned short*)take((size_t)MID_ * COUT_ * 2);
  float* exp_s   = (float*)take(MID_ * 4);
  float* exp_sh  = (float*)take(MID_ * 4);
  float* Ac      = (float*)take(MID_ * 4);
  float* Bc      = (float*)take(MID_ * 4);
  float* Cc      = (float*)take(MID_ * 4);
  float* proj_s  = (float*)take(COUT_ * 4);
  float* proj_sh = (float*)take(COUT_ * 4);
  float* sesum   = (float*)take((size_t)NB * MID_ * 4);
  float* gate    = (float*)take((size_t)NB * MID_ * 4);

  k_prep<<<1, 256, 0, stream>>>(expand_w, expand_bn, dw_bn, id_w, id_bn, merge_bn,
                                proj_w, proj_bn, wb1p, wb2p,
                                exp_s, exp_sh, Ac, Bc, Cc, proj_s, proj_sh);
  k_expand<<<NB * 49, 256, 0, stream>>>(x, wb1p, exp_s, exp_sh, t_bf);
  k_dw<<<NB * MID_, 256, 0, stream>>>(t_bf, dw_w, Ac, Bc, Cc, u_bf, sesum);
  k_se<<<NB, MID_, 0, stream>>>(sesum, se_red_w, se_red_b, se_exp_w, se_exp_b, gate);
  k_project<<<NB * 49, 256, 0, stream>>>(u_bf, gate, wb2p, proj_s, proj_sh, x, out);
}
